// LSTMModel_34213709480092
// MI455X (gfx1250) — matrix operand order repaired and checked by reading, not compiled, not run on hardware
//
#include <hip/hip_runtime.h>
#include <hip/hip_bf16.h>
#include <math.h>

// ---- problem constants (match reference) ----
#define VOCAB 32000
#define EMB   512
#define HID   1024
#define G4H   4096   // 4*HID
#define BATCH 64
#define SEQ   256
#define NCLS  32000

typedef __bf16 bf16_t;
typedef __bf16 v16bf __attribute__((ext_vector_type(16)));
typedef float  v8f   __attribute__((ext_vector_type(8)));
typedef unsigned int u32x4 __attribute__((ext_vector_type(4)));
typedef int          i32x8 __attribute__((ext_vector_type(8)));
typedef int          i32x4 __attribute__((ext_vector_type(4)));

// ---- WMMA tile params ----
#define BM  64
#define BN  128
#define BK  32
#define LDP 40     // padded LDS row stride (bf16): 64B row + 16B pad = 80B

// TDM available? (this toolchain declares the 6-arg clang-23 form)
#if defined(__AMDGCN__) && __has_builtin(__builtin_amdgcn_tensor_load_to_lds)
#define HAS_TDM 1
#else
#define HAS_TDM 0
#endif

// ---------------------------------------------------------------------------
// CDNA5 async global->LDS (per-lane DMA, ASYNCcnt-tracked), 16B per lane.
// ---------------------------------------------------------------------------
__device__ __forceinline__ void async_ld_b128(void* lds_ptr, const void* gptr) {
    const unsigned lds_off = (unsigned)(uintptr_t)lds_ptr;   // low 32b = LDS offset
    const unsigned long long ga = (unsigned long long)(uintptr_t)gptr;
    asm volatile("global_load_async_to_lds_b128 %0, %1, off"
                 :: "v"(lds_off), "v"(ga) : "memory");
}
__device__ __forceinline__ void wait_async_le3() {
    asm volatile("s_wait_asynccnt 0x3" ::: "memory");
}
__device__ __forceinline__ void wait_async_le1() {
    asm volatile("s_wait_asynccnt 0x1" ::: "memory");
}
__device__ __forceinline__ void wait_async_0() {
    asm volatile("s_wait_asynccnt 0x0" ::: "memory");
}

#if HAS_TDM
// ---------------------------------------------------------------------------
// TDM descriptor load: up-to-3D tile of bf16 (data_size=2B) into LDS with
// hardware row padding matching LDP (pad_interval=16 dwords, pad_amount=4).
//   dim0 = contiguous (k), dim1 = rows (stride0), dim2 = gates (stride1).
// All arguments uniform; issue from ONE wave only (EXEC is ignored by TDM).
// 6-arg builtin form: (g0 v4u, g1 v8i, g2 v4i, g3 v4i, extra v8i, cpol).
// ---------------------------------------------------------------------------
__device__ __forceinline__ void tdm_load_bf16(
    unsigned lds_off, const void* gaddr,
    unsigned tile_d0, unsigned tile_d1, unsigned tile_d2,
    unsigned long long stride0_elems, unsigned long long stride1_elems,
    unsigned tens_d0, unsigned tens_d1, unsigned tens_d2)
{
    const unsigned long long ga = (unsigned long long)(uintptr_t)gaddr;
    u32x4 g0;
    g0[0] = 1u;                                   // count=1, user descriptor
    g0[1] = lds_off;                              // LDS byte address
    g0[2] = (unsigned)(ga & 0xFFFFFFFFu);         // global_addr[31:0]
    g0[3] = (unsigned)((ga >> 32) & 0x1FFFFFFu)   // global_addr[56:32]
          | 0x80000000u;                          // type=2 ("image")
    i32x8 g1;
    g1[0] = (int)((1u << 16)                      // data_size: 2 bytes
          |       (1u << 20)                      // pad_enable
          |       (3u << 22)                      // pad_interval: 16 DWORDs
          |       (3u << 25));                    // pad_amount: 4 DWORDs (16B)
    g1[1] = (int)((tens_d0 & 0xFFFFu) << 16);                               // dim0 len lo
    g1[2] = (int)(((tens_d0 >> 16) & 0xFFFFu) | ((tens_d1 & 0xFFFFu) << 16));
    g1[3] = (int)(((tens_d1 >> 16) & 0xFFFFu) | ((tile_d0 & 0xFFFFu) << 16));
    g1[4] = (int)((tile_d1 & 0xFFFFu) | ((tile_d2 & 0xFFFFu) << 16));
    g1[5] = (int)(stride0_elems & 0xFFFFFFFFu);
    g1[6] = (int)(((stride0_elems >> 32) & 0xFFFFu)
          |       ((stride1_elems & 0xFFFFu) << 16));
    g1[7] = (int)((stride1_elems >> 16) & 0xFFFFFFFFu);
    i32x4 g2;
    g2[0] = (int)tens_d2; g2[1] = 0; g2[2] = 0; g2[3] = 0;                  // tensor_dim2
    i32x4 g3; g3[0] = 0; g3[1] = 0; g3[2] = 0; g3[3] = 0;
    i32x8 gx; gx[0] = 0; gx[1] = 0; gx[2] = 0; gx[3] = 0;
              gx[4] = 0; gx[5] = 0; gx[6] = 0; gx[7] = 0;                   // unused words
    __builtin_amdgcn_tensor_load_to_lds(g0, g1, g2, g3, gx, 0);
}
#endif

// ---------------------------------------------------------------------------
// Fragment loaders, per CDNA5 ISA 7.12.2 (wave32):
//   A (16x32 MxK bf16): lane&15 = M row; halves 0..7 -> K = h + 8*(lane>>4),
//                       halves 8..15 -> K = 16 + (h-8) + 8*(lane>>4)
//   B (32x16 KxN bf16), stored N-major in LDS (Bs[n][k]):
//                       lane&15 = N col; same K mapping as A
//   C/D (16x16 f32):    vgpr r, lane l -> M = r + 8*(l>>4), N = l&15
// ---------------------------------------------------------------------------
__device__ __forceinline__ v16bf frag_a(const bf16_t* base, int lane) {
    const bf16_t* p = base + (lane & 15) * LDP + ((lane >> 4) << 3);
    v16bf f;
#pragma unroll
    for (int h = 0; h < 8; ++h) f[h] = p[h];
#pragma unroll
    for (int h = 0; h < 8; ++h) f[8 + h] = p[16 + h];
    return f;
}

__device__ __forceinline__ v16bf frag_b(const bf16_t* base, int lane) {
    const bf16_t* p = base + (lane & 15) * LDP + ((lane >> 4) << 3);
    v16bf f;
#pragma unroll
    for (int h = 0; h < 8; ++h) f[h] = p[h];
#pragma unroll
    for (int h = 0; h < 8; ++h) f[8 + h] = p[16 + h];
    return f;
}

__device__ __forceinline__ v8f wmma_bf16(v16bf a, v16bf b, v8f c) {
    return __builtin_amdgcn_wmma_f32_16x16x32_bf16(
        false, a, false, b, (short)0, c, false, false);
}

// ---------------------------------------------------------------------------
// Generic C = A(MxK) * W(NxK)^T + bias0[n] (+ bias1[n]), bf16 in, f32 out.
// grid = (N/BN, M/BM), block = 256 (8 waves). Double-buffered:
//   A tile via per-lane async-to-LDS, W tile via one 2D TDM descriptor.
// ---------------------------------------------------------------------------
__global__ __launch_bounds__(256)
void k_gemm_bias(const bf16_t* __restrict__ A, const bf16_t* __restrict__ W,
                 const float* __restrict__ bias0, const float* __restrict__ bias1,
                 float* __restrict__ C, int M, int N, int K)
{
    __shared__ __align__(16) bf16_t As[2][BM][LDP];
    __shared__ __align__(16) bf16_t Bs[2][BN][LDP];

    const int n0   = blockIdx.x * BN;
    const int m0   = blockIdx.y * BM;
    const int tid  = threadIdx.x;
    const int lane = tid & 31;
    const int wave = tid >> 5;
    const int tr   = wave & 3;            // C tile row (x16)
    const int tcb  = (wave >> 2) << 2;    // first of 4 C tile cols (x16)

    const int ar = tid >> 2;              // A stage: row 0..63
    const int ac = (tid & 3) << 3;        // A stage: col 0,8,16,24

#if HAS_TDM
    const bool w0 = ((__builtin_amdgcn_readfirstlane((int)tid) >> 5) == 0);
#else
    const int br0 = tid >> 2,          bc0 = (tid & 3) << 3;
    const int br1 = (tid + 256) >> 2,  bc1 = bc0;
#endif

    v8f acc[4] = {};

    // prologue: stage 0
    async_ld_b128(&As[0][ar][ac], A + (size_t)(m0 + ar) * K + ac);
#if HAS_TDM
    if (w0) {
        tdm_load_bf16((unsigned)(uintptr_t)&Bs[0][0][0], W + (size_t)n0 * K,
                      /*tile*/ BK, BN, 0,
                      /*strides*/ (unsigned long long)K, 0ull,
                      /*tensor*/ (unsigned)K, (unsigned)(N - n0), 0);
    }
#else
    async_ld_b128(&Bs[0][br0][bc0], W + (size_t)(n0 + br0) * K + bc0);
    async_ld_b128(&Bs[0][br1][bc1], W + (size_t)(n0 + br1) * K + bc1);
#endif

    int buf = 0;
    for (int kb = 0; kb < K; kb += BK) {
        const bool more = (kb + BK < K);
        if (more) {
            const int nb = buf ^ 1, kn = kb + BK;
            async_ld_b128(&As[nb][ar][ac], A + (size_t)(m0 + ar) * K + kn + ac);
#if HAS_TDM
            if (w0) {
                tdm_load_bf16((unsigned)(uintptr_t)&Bs[nb][0][0],
                              W + (size_t)n0 * K + kn,
                              BK, BN, 0,
                              (unsigned long long)K, 0ull,
                              (unsigned)(K - kn), (unsigned)(N - n0), 0);
            }
#else
            async_ld_b128(&Bs[nb][br0][bc0], W + (size_t)(n0 + br0) * K + kn + bc0);
            async_ld_b128(&Bs[nb][br1][bc1], W + (size_t)(n0 + br1) * K + kn + bc1);
#endif
        }
#if HAS_TDM
        if (more) wait_async_le1(); else wait_async_0();
        if (w0) {
            if (more) __builtin_amdgcn_s_wait_tensorcnt(1);
            else      __builtin_amdgcn_s_wait_tensorcnt(0);
        }
#else
        if (more) wait_async_le3(); else wait_async_0();
#endif
        __syncthreads();

        const v16bf a = frag_a(&As[buf][tr * 16][0], lane);
#pragma unroll
        for (int j = 0; j < 4; ++j) {
            const v16bf b = frag_b(&Bs[buf][(tcb + j) * 16][0], lane);
            acc[j] = wmma_bf16(a, b, acc[j]);
        }
        __syncthreads();          // all waves done reading before buf is re-filled
        buf ^= 1;
    }

    // epilogue: add bias, store fp32
    const int mbase = m0 + tr * 16 + ((lane >> 4) << 3);
    const int nn    = lane & 15;
#pragma unroll
    for (int j = 0; j < 4; ++j) {
        const int gn = n0 + (tcb + j) * 16 + nn;
        float badd = bias0 ? bias0[gn] : 0.0f;
        if (bias1) badd += bias1[gn];
#pragma unroll
        for (int r = 0; r < 8; ++r) {
            C[(size_t)(mbase + r) * N + gn] = acc[j][r] + badd;
        }
    }
}

// ---------------------------------------------------------------------------
// One LSTM timestep. grid = HID/32 blocks; block owns hidden units
// [j0, j0+32), computes all 4 gates for them. The gate-strided Whh tile
// ( [4 gates][32 rows][32 k] at (0, j0, kb) ) is ONE 3D TDM descriptor.
// ---------------------------------------------------------------------------
__global__ __launch_bounds__(256)
void k_lstm_step(const float* __restrict__ XG, const bf16_t* __restrict__ Whh,
                 const bf16_t* __restrict__ hprev, bf16_t* __restrict__ hnext,
                 float* __restrict__ cstate, bf16_t* __restrict__ hseq, int t)
{
    __shared__ __align__(16) bf16_t As[2][BM][LDP];
    __shared__ __align__(16) bf16_t Bs[2][BN][LDP];
    __shared__ float gbuf[BM][BN + 4];

    const int j0   = blockIdx.x * 32;
    const int tid  = threadIdx.x;
    const int lane = tid & 31;
    const int wave = tid >> 5;
    const int tr   = wave & 3;
    const int tcb  = (wave >> 2) << 2;

    const int ar = tid >> 2;
    const int ac = (tid & 3) << 3;

#if HAS_TDM
    const bool w0 = ((__builtin_amdgcn_readfirstlane((int)tid) >> 5) == 0);
#else
    const int br0 = tid >> 2,          bc0 = (tid & 3) << 3;
    const int br1 = (tid + 256) >> 2,  bc1 = bc0;
    const int wr0 = ((br0 >> 5) << 10) + j0 + (br0 & 31);
    const int wr1 = ((br1 >> 5) << 10) + j0 + (br1 & 31);
#endif

    v8f acc[4] = {};

    async_ld_b128(&As[0][ar][ac], hprev + (size_t)ar * HID + ac);
#if HAS_TDM
    if (w0) {
        tdm_load_bf16((unsigned)(uintptr_t)&Bs[0][0][0],
                      Whh + (size_t)j0 * HID,
                      /*tile*/ BK, 32, 4,
                      /*strides*/ (unsigned long long)HID,
                                  (unsigned long long)HID * HID,
                      /*tensor*/ HID, HID, 4);
    }
#else
    async_ld_b128(&Bs[0][br0][bc0], Whh + (size_t)wr0 * HID + bc0);
    async_ld_b128(&Bs[0][br1][bc1], Whh + (size_t)wr1 * HID + bc1);
#endif

    int buf = 0;
    for (int kb = 0; kb < HID; kb += BK) {
        const bool more = (kb + BK < HID);
        if (more) {
            const int nb = buf ^ 1, kn = kb + BK;
            async_ld_b128(&As[nb][ar][ac], hprev + (size_t)ar * HID + kn + ac);
#if HAS_TDM
            if (w0) {
                tdm_load_bf16((unsigned)(uintptr_t)&Bs[nb][0][0],
                              Whh + (size_t)j0 * HID + kn,
                              BK, 32, 4,
                              (unsigned long long)HID,
                              (unsigned long long)HID * HID,
                              (unsigned)(HID - kn), HID, 4);
            }
#else
            async_ld_b128(&Bs[nb][br0][bc0], Whh + (size_t)wr0 * HID + kn + bc0);
            async_ld_b128(&Bs[nb][br1][bc1], Whh + (size_t)wr1 * HID + kn + bc1);
#endif
        }
#if HAS_TDM
        if (more) wait_async_le1(); else wait_async_0();
        if (w0) {
            if (more) __builtin_amdgcn_s_wait_tensorcnt(1);
            else      __builtin_amdgcn_s_wait_tensorcnt(0);
        }
#else
        if (more) wait_async_le3(); else wait_async_0();
#endif
        __syncthreads();

        const v16bf a = frag_a(&As[buf][tr * 16][0], lane);
#pragma unroll
        for (int j = 0; j < 4; ++j) {
            const v16bf b = frag_b(&Bs[buf][(tcb + j) * 16][0], lane);
            acc[j] = wmma_bf16(a, b, acc[j]);
        }
        __syncthreads();
        buf ^= 1;
    }

    // park recurrent contribution in LDS so each thread can gather its 4 gates
    const int mb = tr * 16 + ((lane >> 4) << 3);
    const int nn = lane & 15;
#pragma unroll
    for (int j = 0; j < 4; ++j) {
        const int cn = (tcb + j) * 16 + nn;
#pragma unroll
        for (int r = 0; r < 8; ++r) gbuf[mb + r][cn] = acc[j][r];
    }
    __syncthreads();

    // state update: 64 batch rows x 32 hidden units = 2048 elems, 8/thread
#pragma unroll
    for (int s = 0; s < 8; ++s) {
        const int u  = tid + (s << 8);
        const int b  = u >> 5;
        const int jl = u & 31;
        const size_t xg = ((size_t)b * SEQ + t) * G4H + j0 + jl;
        float gi = gbuf[b][jl]       + XG[xg];
        float gf = gbuf[b][32 + jl]  + XG[xg + HID];
        float gg = gbuf[b][64 + jl]  + XG[xg + 2 * HID];
        float go = gbuf[b][96 + jl]  + XG[xg + 3 * HID];
        gi = 1.0f / (1.0f + __expf(-gi));
        gf = 1.0f / (1.0f + __expf(-gf));
        gg = tanhf(gg);
        go = 1.0f / (1.0f + __expf(-go));
        const int cj = b * HID + j0 + jl;
        const float c = gf * cstate[cj] + gi * gg;
        cstate[cj] = c;
        const float h = go * tanhf(c);
        hnext[cj] = (bf16_t)h;
        if (hseq) hseq[((size_t)b * SEQ + t) * HID + j0 + jl] = (bf16_t)h;
    }
}

// ---------------------------------------------------------------------------
// Embedding gather + fp32 -> bf16
// ---------------------------------------------------------------------------
__global__ void k_embed(const int* __restrict__ tokens,
                        const float* __restrict__ emb, bf16_t* __restrict__ X)
{
    const int bt  = blockIdx.x;            // 0 .. B*T-1
    const int tok = tokens[bt];
    const float* src = emb + (size_t)tok * EMB;
    bf16_t* dst = X + (size_t)bt * EMB;
    for (int c = threadIdx.x; c < EMB; c += blockDim.x) dst[c] = (bf16_t)src[c];
}

__global__ void k_cvt_bf16(const float* __restrict__ src,
                           bf16_t* __restrict__ dst, int n)
{
    int i = blockIdx.x * blockDim.x + threadIdx.x;
    const int stride = gridDim.x * blockDim.x;
    for (; i < n; i += stride) dst[i] = (bf16_t)src[i];
}

__global__ void k_init_state(bf16_t* __restrict__ h0, float* __restrict__ c, int n)
{
    const int i = blockIdx.x * blockDim.x + threadIdx.x;
    if (i < n) { h0[i] = (bf16_t)0.0f; c[i] = 0.0f; }
}

// ---------------------------------------------------------------------------
extern "C" void kernel_launch(void* const* d_in, const int* in_sizes, int n_in,
                              void* d_out, int out_size, void* d_ws, size_t ws_size,
                              hipStream_t stream)
{
    (void)in_sizes; (void)n_in; (void)out_size; (void)ws_size;
    const int*   tokens = (const int*)  d_in[0];
    const float* emb    = (const float*)d_in[1];
    const float* W_ih0  = (const float*)d_in[2];
    const float* W_hh0  = (const float*)d_in[3];
    const float* b_ih0  = (const float*)d_in[4];
    const float* b_hh0  = (const float*)d_in[5];
    const float* W_ih1  = (const float*)d_in[6];
    const float* W_hh1  = (const float*)d_in[7];
    const float* b_ih1  = (const float*)d_in[8];
    const float* b_hh1  = (const float*)d_in[9];
    const float* fc_w   = (const float*)d_in[10];
    const float* fc_b   = (const float*)d_in[11];
    float* out = (float*)d_out;

    // workspace carve-out (256B aligned); total ~405 MB
    char* ws = (char*)d_ws;
    size_t off = 0;
    auto wsalloc = [&](size_t bytes) -> void* {
        void* p = ws + off;
        off += (bytes + 255) & ~((size_t)255);
        return p;
    };
    bf16_t* Wih0b = (bf16_t*)wsalloc((size_t)G4H * EMB * 2);
    bf16_t* Whh0b = (bf16_t*)wsalloc((size_t)G4H * HID * 2);
    bf16_t* Wih1b = (bf16_t*)wsalloc((size_t)G4H * HID * 2);
    bf16_t* Whh1b = (bf16_t*)wsalloc((size_t)G4H * HID * 2);
    bf16_t* fcwb  = (bf16_t*)wsalloc((size_t)NCLS * HID * 2);
    bf16_t* Xb    = (bf16_t*)wsalloc((size_t)BATCH * SEQ * EMB * 2);
    float*  XG    = (float*) wsalloc((size_t)BATCH * SEQ * G4H * 4);
    bf16_t* Hseq  = (bf16_t*)wsalloc((size_t)BATCH * SEQ * HID * 2);
    bf16_t* hb0   = (bf16_t*)wsalloc((size_t)BATCH * HID * 2);
    bf16_t* hb1   = (bf16_t*)wsalloc((size_t)BATCH * HID * 2);
    float*  cst   = (float*) wsalloc((size_t)BATCH * HID * 4);

    // 1) weights -> bf16
    auto cvt = [&](const float* s, bf16_t* d, int n) {
        k_cvt_bf16<<<dim3(2048), dim3(256), 0, stream>>>(s, d, n);
    };
    cvt(W_ih0, Wih0b, G4H * EMB);
    cvt(W_hh0, Whh0b, G4H * HID);
    cvt(W_ih1, Wih1b, G4H * HID);
    cvt(W_hh1, Whh1b, G4H * HID);
    cvt(fc_w,  fcwb,  NCLS * HID);

    // 2) embedding gather
    k_embed<<<dim3(BATCH * SEQ), dim3(256), 0, stream>>>(tokens, emb, Xb);

    // 3) layer 0: XG = X * Wih0^T + b_ih0 + b_hh0   (M=16384, N=4096, K=512)
    k_gemm_bias<<<dim3(G4H / BN, (BATCH * SEQ) / BM), dim3(256), 0, stream>>>(
        Xb, Wih0b, b_ih0, b_hh0, XG, BATCH * SEQ, G4H, EMB);

    // 4) layer 0 recurrence (256 sequential steps, ping-pong h buffers)
    k_init_state<<<dim3((BATCH * HID + 255) / 256), dim3(256), 0, stream>>>(
        hb0, cst, BATCH * HID);
    for (int t = 0; t < SEQ; ++t) {
        bf16_t* hp = (t & 1) ? hb1 : hb0;
        bf16_t* hn = (t & 1) ? hb0 : hb1;
        k_lstm_step<<<dim3(HID / 32), dim3(256), 0, stream>>>(
            XG, Whh0b, hp, hn, cst, Hseq, t);
    }

    // 5) layer 1: XG = Hseq * Wih1^T + b_ih1 + b_hh1 (M=16384, N=4096, K=1024)
    k_gemm_bias<<<dim3(G4H / BN, (BATCH * SEQ) / BM), dim3(256), 0, stream>>>(
        Hseq, Wih1b, b_ih1, b_hh1, XG, BATCH * SEQ, G4H, HID);

    // 6) layer 1 recurrence
    k_init_state<<<dim3((BATCH * HID + 255) / 256), dim3(256), 0, stream>>>(
        hb0, cst, BATCH * HID);
    for (int t = 0; t < SEQ; ++t) {
        bf16_t* hp = (t & 1) ? hb1 : hb0;
        bf16_t* hn = (t & 1) ? hb0 : hb1;
        k_lstm_step<<<dim3(HID / 32), dim3(256), 0, stream>>>(
            XG, Whh1b, hp, hn, cst, nullptr, t);
    }
    // t=255 (odd) wrote hb0 -> final hidden state

    // 7) FC: out = h_last * fc_w^T + fc_b   (M=64, N=32000, K=1024)
    k_gemm_bias<<<dim3(NCLS / BN, BATCH / BM), dim3(256), 0, stream>>>(
        hb0, fcwb, fc_b, nullptr, out, BATCH, NCLS, HID);
}
